// VSS3DBottleneck_28475633173012
// MI455X (gfx1250) — matrix operand order repaired and checked by reading, not compiled, not run on hardware
//
#include <hip/hip_runtime.h>
#include <hip/hip_bf16.h>
#include <math.h>

// ---------------------------------------------------------------------------
// VSS3D bottleneck for MI455X (gfx1250, wave32, WMMA)
//   B=2, C=256, D=H=W=16 -> L=4096, M = B*L = 8192
//   GEMMs run as f16 WMMA (v_wmma_f32_16x16x32_f16) with fp32 accumulate.
//   Global->LDS staging uses CDNA5 async-to-LDS (ASYNCcnt) when available.
// ---------------------------------------------------------------------------

#define BATCH 2
#define CDIM  256
#define SDIM  16            // spatial edge
#define LLEN  4096          // 16*16*16
#define MROWS 8192          // BATCH * LLEN
#define DSN   16            // state dim
#define DTR   16            // dt rank
#define DBLN  48            // DTR + 2*DSN

typedef _Float16 h8   __attribute__((ext_vector_type(8)));
typedef _Float16 v16h __attribute__((ext_vector_type(16)));
typedef float    v8f  __attribute__((ext_vector_type(8)));

#if defined(__HIP_DEVICE_COMPILE__) &&                                        \
    __has_builtin(__builtin_amdgcn_global_load_async_to_lds_b128) &&          \
    __has_builtin(__builtin_amdgcn_s_wait_asynccnt)
#define USE_ASYNC_LDS 1
// payload type per the builtin's signature: int __attribute__((vector_size(16)))
typedef int v4i_b128 __attribute__((vector_size(16)));
typedef __attribute__((address_space(1))) v4i_b128 gbl_b128;   // global src
typedef __attribute__((address_space(3))) v4i_b128 lds_b128;   // LDS dst
#else
#define USE_ASYNC_LDS 0
#endif

// ---------------------------------------------------------------------------
// Generic WMMA GEMM:  Out[M,N](f32) = A[M,K](f16,row-major) * W[N,K]^T (f16)
// Optional residual add (Res may alias Out; each element RMW'd by one lane).
// Block = 256 threads = 8 waves. Tile: 64(M) x 64(N), K stepped by 32.
// Wave w: wm = w&3 (M subtile), wn = w>>2 (32-wide N chunk -> two 16x16 tiles)
// ---------------------------------------------------------------------------
__global__ __launch_bounds__(256) void gemm_wmma_f16(
    const _Float16* __restrict__ A, const _Float16* __restrict__ W,
    const float* Res, float* Out, int M, int N, int K, int addRes)
{
    __shared__ __align__(16) _Float16 As[64 * 48];   // 64 rows x 32 (stride 48)
    __shared__ __align__(16) _Float16 Bs[64 * 48];   // Bs[n][k] (transposed W)

    const int tid  = threadIdx.x;
    const int wave = tid >> 5;
    const int lane = tid & 31;
    const int row0 = blockIdx.x * 64;
    const int col0 = blockIdx.y * 64;
    const int wm   = wave & 3;
    const int wn   = wave >> 2;

    const int lm   = lane & 15;
    const int kbA  = (lane < 16) ? 0 : 16;   // 16-bit A fragment K base
    const int kbB  = (lane < 16) ? 0 : 16;   // 16-bit B fragment K base
    const int ldr  = tid >> 2;               // 0..63  (row / col for staging)
    const int ldk  = (tid & 3) * 8;          // 0,8,16,24

    const bool fullB = (col0 + 64 <= N);     // block-uniform

    v8f acc0 = {}; v8f acc1 = {};

    for (int k0 = 0; k0 < K; k0 += 32) {
        const _Float16* gA = A + (size_t)(row0 + ldr) * K + k0 + ldk;
        _Float16*       sA = &As[ldr * 48 + ldk];
        const _Float16* gB = W + (size_t)(col0 + ldr) * K + k0 + ldk;
        _Float16*       sB = &Bs[ldr * 48 + ldk];

#if USE_ASYNC_LDS
        // ---- CDNA5 async global->LDS staging (no VGPR bounce) ----
        __builtin_amdgcn_global_load_async_to_lds_b128(
            (gbl_b128*)gA, (lds_b128*)sA, 0, 0);
        if (fullB) {
            __builtin_amdgcn_global_load_async_to_lds_b128(
                (gbl_b128*)gB, (lds_b128*)sB, 0, 0);
        } else {
            h8 bvec = {};
            if (col0 + ldr < N) bvec = *(const h8*)gB;
            *(h8*)sB = bvec;
        }
        if (k0 + 32 < K)
            __builtin_prefetch(gA + 32, 0, 1);
        __builtin_amdgcn_s_wait_asynccnt(0);
#else
        *(h8*)sA = *(const h8*)gA;
        h8 bvec = {};
        if (fullB || (col0 + ldr < N)) bvec = *(const h8*)gB;
        *(h8*)sB = bvec;
        if (k0 + 32 < K)
            __builtin_prefetch(gA + 32, 0, 1);
#endif
        __syncthreads();

        // ---- build A fragment: lane holds M=wm*16+lm, K = kbA..kbA+15 ----
        v16h af;
        {
            const h8 alo = *(const h8*)(&As[(wm * 16 + lm) * 48 + kbA]);
            const h8 ahi = *(const h8*)(&As[(wm * 16 + lm) * 48 + kbA + 8]);
            #pragma unroll
            for (int i = 0; i < 8; ++i) { af[i] = alo[i]; af[8 + i] = ahi[i]; }
        }
        // ---- build B fragments: lane holds N=lm(+16), K = kbB..kbB+15 ----
        v16h b0, b1;
        {
            const int n0 = wn * 32 + lm;
            const h8 x0 = *(const h8*)(&Bs[n0 * 48 + kbB]);
            const h8 x1 = *(const h8*)(&Bs[n0 * 48 + kbB + 8]);
            const int n1 = n0 + 16;
            const h8 y0 = *(const h8*)(&Bs[n1 * 48 + kbB]);
            const h8 y1 = *(const h8*)(&Bs[n1 * 48 + kbB + 8]);
            #pragma unroll
            for (int i = 0; i < 8; ++i) {
                b0[i] = x0[i]; b0[8 + i] = x1[i];
                b1[i] = y0[i]; b1[8 + i] = y1[i];
            }
        }
        acc0 = __builtin_amdgcn_wmma_f32_16x16x32_f16(false, af, false, b0,
                                                      (short)0, acc0, false, false);
        acc1 = __builtin_amdgcn_wmma_f32_16x16x32_f16(false, af, false, b1,
                                                      (short)0, acc1, false, false);
        __syncthreads();
    }

    // ---- store (C/D layout: lane n=lm, VGPR v -> M = v + (lane<16?0:8)) ----
    const int mb  = (lane < 16) ? 0 : 8;
    const int gr  = row0 + wm * 16 + mb;
    const int gc0 = col0 + wn * 32 + lm;
    const int gc1 = gc0 + 16;
    #pragma unroll
    for (int v = 0; v < 8; ++v) {
        const int r = gr + v;
        if (gc0 < N) {
            float val = acc0[v];
            if (addRes) val += Res[(size_t)r * N + gc0];
            Out[(size_t)r * N + gc0] = val;
        }
        if (gc1 < N) {
            float val = acc1[v];
            if (addRes) val += Res[(size_t)r * N + gc1];
            Out[(size_t)r * N + gc1] = val;
        }
    }
}

// ---------------------------------------------------------------------------
// Helpers / elementwise kernels
// ---------------------------------------------------------------------------
__device__ __forceinline__ float blockSum256(float v, float* red) {
    const int c = threadIdx.x;
    red[c] = v; __syncthreads();
    #pragma unroll
    for (int s = 128; s > 0; s >>= 1) {
        if (c < s) red[c] += red[c + s];
        __syncthreads();
    }
    const float r = red[0];
    __syncthreads();
    return r;
}

// t[m][c] = x[b][c][d][h][w]   with m = ((b*16+h)*16+w)*16+d
__global__ __launch_bounds__(256) void transpose_in_k(
    const float* __restrict__ x, float* __restrict__ t)
{
    const int m = blockIdx.x, c = threadIdx.x;
    const int d = m & 15, w = (m >> 4) & 15, h = (m >> 8) & 15, b = m >> 12;
    t[(size_t)m * CDIM + c] =
        x[((((size_t)b * CDIM + c) * SDIM + d) * SDIM + h) * SDIM + w];
}

__global__ void cvt_f32_to_f16_k(const float* __restrict__ in,
                                 _Float16* __restrict__ out, int n)
{
    for (int i = blockIdx.x * blockDim.x + threadIdx.x; i < n;
         i += gridDim.x * blockDim.x)
        out[i] = (_Float16)in[i];
}

// LayerNorm over C, emit f16 activation
__global__ __launch_bounds__(256) void ln_f16_k(
    const float* __restrict__ X, const float* __restrict__ g,
    const float* __restrict__ b, _Float16* __restrict__ O)
{
    __shared__ float red[256];
    const int m = blockIdx.x, c = threadIdx.x;
    const float v = X[(size_t)m * CDIM + c];
    const float mean = blockSum256(v, red) * (1.f / CDIM);
    const float dv = v - mean;
    const float var = blockSum256(dv * dv, red) * (1.f / CDIM);
    O[(size_t)m * CDIM + c] =
        (_Float16)(dv * rsqrtf(var + 1e-5f) * g[c] + b[c]);
}

// depthwise 3x3x3 conv over (h,w,d) on xs = xz[:, :256], SiLU, flip -> seq16
__global__ __launch_bounds__(256) void conv_silu_flip_k(
    const float* __restrict__ xz, const float* __restrict__ cw,
    const float* __restrict__ cb, _Float16* __restrict__ seq16, int ori)
{
    const int m = blockIdx.x, c = threadIdx.x;
    const int d = m & 15, w = (m >> 4) & 15, h = (m >> 8) & 15, b = m >> 12;
    float s = cb[c];
    #pragma unroll
    for (int i = 0; i < 3; ++i) {
        const int hh = h + i - 1; if ((unsigned)hh >= 16u) continue;
        #pragma unroll
        for (int j = 0; j < 3; ++j) {
            const int ww = w + j - 1; if ((unsigned)ww >= 16u) continue;
            #pragma unroll
            for (int k = 0; k < 3; ++k) {
                const int dd = d + k - 1; if ((unsigned)dd >= 16u) continue;
                const int mm = (((b * 16 + hh) * 16 + ww) * 16 + dd);
                s += xz[(size_t)mm * 512 + c] * cw[((c * 3 + i) * 3 + j) * 3 + k];
            }
        }
    }
    s = s / (1.f + __expf(-s));                         // SiLU
    const int hf = (ori & 1) ? 15 - h : h;
    const int wf = (ori & 2) ? 15 - w : w;
    const int df = (ori & 4) ? 15 - d : d;
    const int lf = (hf * 16 + wf) * 16 + df;
    seq16[((size_t)(b * LLEN + lf)) * CDIM + c] = (_Float16)s;
}

// dt[m][c] = softplus( sum_r dbl[m][r] * dtw[c][r] + dtb[c] ),  r < 16
__global__ __launch_bounds__(256) void dt_softplus_k(
    const float* __restrict__ dbl, const float* __restrict__ dtw,
    const float* __restrict__ dtb, float* __restrict__ dt)
{
    __shared__ float dr[DTR];
    const int m = blockIdx.x, c = threadIdx.x;
    if (c < DTR) dr[c] = dbl[(size_t)m * DBLN + c];
    __syncthreads();
    float s = dtb[c];
    #pragma unroll
    for (int r = 0; r < DTR; ++r) s += dr[r] * dtw[c * DTR + r];
    dt[(size_t)m * CDIM + c] = (s > 20.f) ? s : log1pf(__expf(s));
}

// selective scan: one block per batch, thread = channel, 16 states in regs
__global__ __launch_bounds__(256) void scan_k(
    const float* __restrict__ dt, const _Float16* __restrict__ seq16,
    const float* __restrict__ dbl, const float* __restrict__ A_log,
    const float* __restrict__ Dsk, float* __restrict__ y)
{
    __shared__ float BC[2 * DSN];        // Bm (16) then Cm (16) for current l
    const int b = blockIdx.x, c = threadIdx.x;
    float Aneg[DSN], carry[DSN];
    #pragma unroll
    for (int n = 0; n < DSN; ++n) {
        Aneg[n] = -__expf(A_log[c * DSN + n]);
        carry[n] = 0.f;
    }
    const float Dc = Dsk[c];
    for (int l = 0; l < LLEN; ++l) {
        const size_t mrow = (size_t)(b * LLEN + l);
        if (c < 2 * DSN) BC[c] = dbl[mrow * DBLN + DTR + c];
        __syncthreads();
        const float dtv = dt[mrow * CDIM + c];
        const float sv  = (float)seq16[mrow * CDIM + c];
        const float dts = dtv * sv;
        float acc = 0.f;
        #pragma unroll
        for (int n = 0; n < DSN; ++n) {
            const float dA = __expf(dtv * Aneg[n]);
            carry[n] = dA * carry[n] + dts * BC[n];
            acc += carry[n] * BC[DSN + n];
        }
        y[mrow * CDIM + c] = acc + Dc * sv;
        __syncthreads();
    }
}

// unflip + output LayerNorm + SiLU(z) gate -> f16 for the out-proj GEMM
__global__ __launch_bounds__(256) void gate_ln_k(
    const float* __restrict__ y, const float* __restrict__ xz,
    const float* __restrict__ og, const float* __restrict__ ob,
    _Float16* __restrict__ g16, int ori)
{
    __shared__ float red[256];
    const int m = blockIdx.x, c = threadIdx.x;
    const int d = m & 15, w = (m >> 4) & 15, h = (m >> 8) & 15, b = m >> 12;
    const int hf = (ori & 1) ? 15 - h : h;
    const int wf = (ori & 2) ? 15 - w : w;
    const int df = (ori & 4) ? 15 - d : d;
    const size_t mrow = (size_t)(b * LLEN + ((hf * 16 + wf) * 16 + df));
    const float v = y[mrow * CDIM + c];
    const float mean = blockSum256(v, red) * (1.f / CDIM);
    const float dv = v - mean;
    const float var = blockSum256(dv * dv, red) * (1.f / CDIM);
    float o = dv * rsqrtf(var + 1e-5f) * og[c] + ob[c];
    const float z = xz[(size_t)m * 512 + CDIM + c];
    o *= z / (1.f + __expf(-z));                        // * SiLU(z)
    g16[(size_t)m * CDIM + c] = (_Float16)o;
}

// final LayerNorm + transpose back to (B, C, D, H, W)
__global__ __launch_bounds__(256) void final_ln_k(
    const float* __restrict__ t, const float* __restrict__ pg,
    const float* __restrict__ pb, float* __restrict__ out)
{
    __shared__ float red[256];
    const int m = blockIdx.x, c = threadIdx.x;
    const float v = t[(size_t)m * CDIM + c];
    const float mean = blockSum256(v, red) * (1.f / CDIM);
    const float dv = v - mean;
    const float var = blockSum256(dv * dv, red) * (1.f / CDIM);
    const float o = dv * rsqrtf(var + 1e-5f) * pg[c] + pb[c];
    const int d = m & 15, w = (m >> 4) & 15, h = (m >> 8) & 15, b = m >> 12;
    out[((((size_t)b * CDIM + c) * SDIM + d) * SDIM + h) * SDIM + w] = o;
}

// ---------------------------------------------------------------------------
extern "C" void kernel_launch(void* const* d_in, const int* in_sizes, int n_in,
                              void* d_out, int out_size, void* d_ws, size_t ws_size,
                              hipStream_t stream)
{
    const float* x      = (const float*)d_in[0];
    const float* in_w   = (const float*)d_in[1];
    const float* conv_w = (const float*)d_in[2];
    const float* conv_b = (const float*)d_in[3];
    const float* xproj  = (const float*)d_in[4];
    const float* dt_w   = (const float*)d_in[5];
    const float* dt_b   = (const float*)d_in[6];
    const float* A_log  = (const float*)d_in[7];
    const float* D_skip = (const float*)d_in[8];
    const float* on_g   = (const float*)d_in[9];
    const float* on_b   = (const float*)d_in[10];
    const float* out_w  = (const float*)d_in[11];
    const float* ln1_g  = (const float*)d_in[12];
    const float* ln1_b  = (const float*)d_in[13];
    const float* post_g = (const float*)d_in[14];
    const float* post_b = (const float*)d_in[15];
    float* out = (float*)d_out;

    // workspace carve-up (256B aligned)
    size_t off = 0;
    auto carve = [&](size_t bytes) -> void* {
        void* p = (char*)d_ws + off;
        off += (bytes + 255) & ~(size_t)255;
        return p;
    };
    float*    t      = (float*)   carve((size_t)MROWS * CDIM * 4);
    _Float16* h16    = (_Float16*)carve((size_t)MROWS * CDIM * 2);
    float*    xz     = (float*)   carve((size_t)MROWS * 512  * 4);
    _Float16* seq16  = (_Float16*)carve((size_t)MROWS * CDIM * 2);
    float*    dbl    = (float*)   carve((size_t)MROWS * DBLN * 4);
    float*    dtb    = (float*)   carve((size_t)MROWS * CDIM * 4);
    float*    yb     = (float*)   carve((size_t)MROWS * CDIM * 4);
    _Float16* g16    = (_Float16*)carve((size_t)MROWS * CDIM * 2);
    _Float16* w16in  = (_Float16*)carve((size_t)4 * 512 * CDIM * 2);
    _Float16* w16xp  = (_Float16*)carve((size_t)4 * DBLN * CDIM * 2);
    _Float16* w16out = (_Float16*)carve((size_t)4 * CDIM * CDIM * 2);
    (void)ws_size; (void)n_in; (void)in_sizes; (void)out_size;

    // input transpose + one-time weight conversions to f16
    transpose_in_k<<<MROWS, 256, 0, stream>>>(x, t);
    cvt_f32_to_f16_k<<<512, 256, 0, stream>>>(in_w,  w16in,  4 * 512 * CDIM);
    cvt_f32_to_f16_k<<<64,  256, 0, stream>>>(xproj, w16xp,  4 * DBLN * CDIM);
    cvt_f32_to_f16_k<<<256, 256, 0, stream>>>(out_w, w16out, 4 * CDIM * CDIM);

    for (int i = 0; i < 4; ++i) {
        const int ori = i & 7;
        // 1) LN1 -> f16
        ln_f16_k<<<MROWS, 256, 0, stream>>>(t, ln1_g + i * CDIM, ln1_b + i * CDIM, h16);
        // 2) xz = h @ in_w^T   (M=8192, N=512, K=256)
        gemm_wmma_f16<<<dim3(MROWS / 64, 512 / 64), 256, 0, stream>>>(
            h16, w16in + (size_t)i * 512 * CDIM, nullptr, xz, MROWS, 512, CDIM, 0);
        // 3) depthwise conv + SiLU + orientation flip -> seq16
        conv_silu_flip_k<<<MROWS, 256, 0, stream>>>(
            xz, conv_w + (size_t)i * CDIM * 27, conv_b + i * CDIM, seq16, ori);
        // 4) dbl = seq @ xproj^T  (N=48, edge-guarded)
        gemm_wmma_f16<<<dim3(MROWS / 64, 1), 256, 0, stream>>>(
            seq16, w16xp + (size_t)i * DBLN * CDIM, nullptr, dbl, MROWS, DBLN, CDIM, 0);
        // 5) dt = softplus(dbl[:,:16] @ dt_w^T + dt_b)
        dt_softplus_k<<<MROWS, 256, 0, stream>>>(
            dbl, dt_w + (size_t)i * CDIM * DTR, dt_b + i * CDIM, dtb);
        // 6) sequential selective scan (one block per batch)
        scan_k<<<BATCH, 256, 0, stream>>>(
            dtb, seq16, dbl, A_log + (size_t)i * CDIM * DSN, D_skip + i * CDIM, yb);
        // 7) unflip + out-LN + SiLU(z) gate -> f16
        gate_ln_k<<<MROWS, 256, 0, stream>>>(
            yb, xz, on_g + i * CDIM, on_b + i * CDIM, g16, ori);
        // 8) t += g @ out_w^T  (fused residual)
        gemm_wmma_f16<<<dim3(MROWS / 64, CDIM / 64), 256, 0, stream>>>(
            g16, w16out + (size_t)i * CDIM * CDIM, t, t, MROWS, CDIM, CDIM, 1);
    }

    // final LN + transpose to (B, C, D, H, W)
    final_ln_k<<<MROWS, 256, 0, stream>>>(t, post_g, post_b, out);
}
